// BeliefEncoder_49048526520411
// MI455X (gfx1250) — matrix operand order repaired and checked by reading, not compiled, not run on hardware
//
#include <hip/hip_runtime.h>
#include <hip/hip_bf16.h>
#include <cstdint>

// LSTM (flax OptimizedLSTMCell) fused persistent kernel for MI455X (gfx1250).
// B=128, T=1024, D=512, H=512.  y = [x_t, h_t] @ Wc^T + b, gates -> c,h.
// Wc^T is a bf16 (N=4H=2048, K=D+H=1024) row-major copy built once in d_ws.
// x_{t+1} is prefetched into registers during step t's WMMA K-loop so HBM
// latency never sits on the sequential recurrence chain.
// LDS x|h panel rows padded to 1032 ushorts (516 dwords == 4 mod 64 banks)
// so the per-k-step ds_load_b128 A-fragment reads are minimally conflicting.

typedef __attribute__((ext_vector_type(16))) __bf16 v16bf;
typedef __attribute__((ext_vector_type(8)))  float  v8f;

static constexpr int Bb = 128, Tt = 1024, Dd = 512, Hh = 512;
static constexpr int Kk = Dd + Hh;      // 1024
static constexpr int Nn = 4 * Hh;       // 2048
static constexpr int KkP = Kk + 8;      // padded LDS row: 1032 ushorts = 516 dwords
static constexpr int CTX = Bb * Tt * Hh;

__device__ __forceinline__ unsigned short f2bf(float f) {
  unsigned int u = __builtin_bit_cast(unsigned int, f);
  unsigned int r = u + 0x7FFFu + ((u >> 16) & 1u);   // round-to-nearest-even
  return (unsigned short)(r >> 16);
}

__device__ __forceinline__ float sigf(float x) {
  return __builtin_amdgcn_rcpf(1.0f + __expf(-x));
}
__device__ __forceinline__ float tanh_fast(float x) {
  float e = __expf(-2.0f * x);
  return (1.0f - e) * __builtin_amdgcn_rcpf(1.0f + e);
}

struct alignas(16) BF16x16 { uint4 lo; uint4 hi; };  // 32B -> v16bf

// Build Wc^T bf16: Wt[n][k] = (k < D ? Wi[k][n] : Wh[k-D][n])
__global__ void prep_weights(const float* __restrict__ Wi,
                             const float* __restrict__ Wh,
                             unsigned short* __restrict__ Wt) {
  int idx = blockIdx.x * 256 + threadIdx.x;   // n*Kk + k
  int n = idx >> 10;
  int k = idx & (Kk - 1);
  float v = (k < Dd) ? Wi[k * Nn + n] : Wh[(k - Dd) * Nn + n];
  Wt[idx] = f2bf(v);
}

__launch_bounds__(512, 1)
__global__ void lstm_fused(const float* __restrict__ obs,            // (B,T,D) f32
                           const unsigned short* __restrict__ Wt,    // (N,K) bf16
                           const float* __restrict__ bias,           // (N)
                           const float* __restrict__ c0,             // (B,H)
                           const float* __restrict__ h0,             // (B,H)
                           float* __restrict__ out) {                // ctx + cT + hT
  __shared__ __align__(16) unsigned short xh[16][KkP];  // ~33 KB: [x_t | h_t] bf16

  const int tid  = threadIdx.x;
  const int wave = tid >> 5;      // 0..15 : owns hidden cols [wave*32, wave*32+32)
  const int lane = tid & 31;
  const int l15  = lane & 15;     // B-operand column / A-operand row
  const int lhi  = lane >> 4;     // K-half select (WMMA VGPR layout)
  const int m0   = blockIdx.x * 16;  // batch rows [m0, m0+16)

  // Per-(gate g, sub-tile u) weight offsets and bias (cols n = g*H + wave*32 + u*16 + l15)
  int   boff[4][2];
  float brow[4][2];
  #pragma unroll
  for (int g = 0; g < 4; ++g)
    #pragma unroll
    for (int u = 0; u < 2; ++u) {
      int n = g * Hh + wave * 32 + u * 16 + l15;
      boff[g][u] = n * Kk + lhi * 8;
      brow[g][u] = bias[n];
    }

  // Cell state lives in VGPRs for all T steps; h fed back via LDS as bf16.
  v8f cst[2];
  unsigned short hprev[2][8];
  #pragma unroll
  for (int u = 0; u < 2; ++u) {
    int j = wave * 32 + u * 16 + l15;
    #pragma unroll
    for (int r = 0; r < 8; ++r) {
      int m = r + 8 * lhi;
      cst[u][r]   = c0[(m0 + m) * Hh + j];
      hprev[u][r] = f2bf(h0[(m0 + m) * Hh + j]);
    }
  }

  // Prime the x double-buffer: each thread owns 4 float4 chunks of the 16xD tile.
  // chunk c -> i = c*512 + tid ; m = i>>7 ; c4 = i&127
  float4 xreg[4];
  #pragma unroll
  for (int c = 0; c < 4; ++c) {
    int i = c * 512 + tid;
    int m = i >> 7, c4 = i & 127;
    xreg[c] = *(const float4*)&obs[((m0 + m) * Tt + 0) * Dd + c4 * 4];
  }

  for (int t = 0; t < Tt; ++t) {
    // ---- stage h_t (previous output / h0) into LDS [D..K)
    #pragma unroll
    for (int u = 0; u < 2; ++u) {
      int j = wave * 32 + u * 16 + l15;
      #pragma unroll
      for (int r = 0; r < 8; ++r)
        xh[r + 8 * lhi][Dd + j] = hprev[u][r];
    }
    // ---- stage x_t from registers into LDS [0..D) as bf16
    #pragma unroll
    for (int c = 0; c < 4; ++c) {
      int i = c * 512 + tid;
      int m = i >> 7, c4 = i & 127;
      ushort4 p;
      p.x = f2bf(xreg[c].x); p.y = f2bf(xreg[c].y);
      p.z = f2bf(xreg[c].z); p.w = f2bf(xreg[c].w);
      *(ushort4*)&xh[m][c4 * 4] = p;
    }
    // ---- issue x_{t+1} loads now; they complete under the WMMA K-loop
    {
      int tn = (t + 1 < Tt) ? t + 1 : t;
      #pragma unroll
      for (int c = 0; c < 4; ++c) {
        int i = c * 512 + tid;
        int m = i >> 7, c4 = i & 127;
        xreg[c] = *(const float4*)&obs[((m0 + m) * Tt + tn) * Dd + c4 * 4];
      }
    }
    __syncthreads();

    // ---- y = [x,h] @ Wc^T + b : 8 accumulator tiles (4 gates x 2), K=1024
    v8f acc[4][2];
    #pragma unroll
    for (int g = 0; g < 4; ++g)
      #pragma unroll
      for (int u = 0; u < 2; ++u)
        #pragma unroll
        for (int r = 0; r < 8; ++r)
          acc[g][u][r] = brow[g][u];

    #pragma unroll 4
    for (int kk = 0; kk < Kk / 32; ++kk) {
      // A fragment: row l15, K = kk*32 + {lhi*8..+8} and {16+lhi*8..+8}
      BF16x16 ap;
      ap.lo = *(const uint4*)&xh[l15][kk * 32 + lhi * 8];
      ap.hi = *(const uint4*)&xh[l15][kk * 32 + 16 + lhi * 8];
      const v16bf a = __builtin_bit_cast(v16bf, ap);
      #pragma unroll
      for (int g = 0; g < 4; ++g)
        #pragma unroll
        for (int u = 0; u < 2; ++u) {
          // B fragment: column n, K = kk*32 + {lhi*8..+8} and {16+lhi*8..+8} (L2-resident weights)
          const uint4* q = (const uint4*)(Wt + boff[g][u] + kk * 32);
          BF16x16 bp; bp.lo = q[0]; bp.hi = q[2];
          const v16bf b = __builtin_bit_cast(v16bf, bp);
          acc[g][u] = __builtin_amdgcn_wmma_f32_16x16x32_bf16(
              false, a, false, b, (short)0, acc[g][u], false, false);
        }
    }

    // ---- gates, state update, output store
    #pragma unroll
    for (int u = 0; u < 2; ++u) {
      int j = wave * 32 + u * 16 + l15;
      #pragma unroll
      for (int r = 0; r < 8; ++r) {
        int m = r + 8 * lhi;
        float iv = sigf(acc[0][u][r]);
        float fv = sigf(acc[1][u][r]);
        float gv = tanh_fast(acc[2][u][r]);
        float ov = sigf(acc[3][u][r]);
        float cn = fv * cst[u][r] + iv * gv;
        float hn = ov * tanh_fast(cn);
        cst[u][r]   = cn;
        hprev[u][r] = f2bf(hn);
        out[((m0 + m) * Tt + t) * Hh + j] = hn;       // contexts (B,T,H)
        if (t == Tt - 1) {
          out[CTX + (m0 + m) * Hh + j]           = cn; // cT
          out[CTX + Bb * Hh + (m0 + m) * Hh + j] = hn; // hT
        }
      }
    }
    __syncthreads();
  }
}

extern "C" void kernel_launch(void* const* d_in, const int* in_sizes, int n_in,
                              void* d_out, int out_size, void* d_ws, size_t ws_size,
                              hipStream_t stream) {
  (void)in_sizes; (void)n_in; (void)out_size; (void)ws_size;
  const float* obs = (const float*)d_in[0];
  const float* Wi  = (const float*)d_in[1];
  const float* Wh  = (const float*)d_in[2];
  const float* b   = (const float*)d_in[3];
  const float* c0  = (const float*)d_in[4];
  const float* h0  = (const float*)d_in[5];
  float* out = (float*)d_out;
  unsigned short* Wt = (unsigned short*)d_ws;   // 2048*1024*2 = 4 MB bf16

  prep_weights<<<(Nn * Kk) / 256, 256, 0, stream>>>(Wi, Wh, Wt);
  lstm_fused<<<Bb / 16, 512, 0, stream>>>(obs, Wt, b, c0, h0, out);
}
